// GCN_vanilla_3_layers_31593779430027
// MI455X (gfx1250) — matrix operand order repaired and checked by reading, not compiled, not run on hardware
//
#include <hip/hip_runtime.h>

typedef __attribute__((ext_vector_type(16))) _Float16 v16h;
typedef __attribute__((ext_vector_type(8)))  _Float16 v8h;
typedef __attribute__((ext_vector_type(4)))  _Float16 v4h;
typedef __attribute__((ext_vector_type(8)))  float    v8f;
typedef __attribute__((ext_vector_type(4)))  float    vf4;

#define N_NODES 50000
#define N_EDGES 300000
#define NFEAT   512
#define HID1    256
#define HID2    128
#define NHID    64

// ---------------- workspace layout (bytes, 256-aligned regions) ----------------
static constexpr size_t SZ_A     = (size_t)N_NODES * NFEAT * 2;   // 51,200,000  f16 activations
static constexpr size_t SZ_B     = (size_t)N_NODES * HID1  * 4;   // 51,200,000  f32 support
static constexpr size_t SZ_OFFS  = 50048 * 4;                     // 200,192 (padded 50001 ints)
static constexpr size_t SZ_CSR   = 1200128;                       // padded 300000*4

static constexpr size_t OFF_A    = 0;                             // xh -> h1h -> h2h
static constexpr size_t OFF_B    = OFF_A   + SZ_A;                // support (GEMM out)
static constexpr size_t OFF_OFFS = OFF_B   + SZ_B;                // CSR row offsets [N+1]
static constexpr size_t OFF_CNT  = OFF_OFFS + SZ_OFFS;            // hist counts -> fill cursor
static constexpr size_t OFF_CSRS = OFF_CNT + SZ_OFFS;             // CSR src indices [E]
static constexpr size_t OFF_CSRV = OFF_CSRS + SZ_CSR;             // CSR edge values [E]
static constexpr size_t OFF_W1   = OFF_CSRV + SZ_CSR;             // packed f16 weights
static constexpr size_t OFF_W2   = OFF_W1 + 262144;
static constexpr size_t OFF_W3   = OFF_W2 + 65536;

// ---------------- small utility kernels ----------------
__global__ void cast_f32_to_f16_x4(const float* __restrict__ s,
                                   _Float16* __restrict__ d, int n4) {
  int i = blockIdx.x * blockDim.x + threadIdx.x;
  if (i >= n4) return;
  vf4 v = ((const vf4*)s)[i];
  v4h h;
  h.x = (_Float16)v.x; h.y = (_Float16)v.y;
  h.z = (_Float16)v.z; h.w = (_Float16)v.w;
  ((v4h*)d)[i] = h;
}

__global__ void zero_f32_x4(float* __restrict__ p, int n4) {
  int i = blockIdx.x * blockDim.x + threadIdx.x;
  if (i >= n4) return;
  vf4 z = {0.0f, 0.0f, 0.0f, 0.0f};
  ((vf4*)p)[i] = z;
}

// Pack W [K][NOUT] (f32, row major) into per-lane WMMA B fragments (f16).
// Fragment (kt,nt): 32 lanes x 16 halfs; lane l holds column n = nt*16 + (l&15),
// K = kt*32 + ((l&16)?8:0) + (i&7) + ((i&8)?16:0)  for i in 0..15 (table order).
__global__ void pack_w(const float* __restrict__ W, _Float16* __restrict__ Wp,
                       int NT, int NOUT, int total) {
  int idx = blockIdx.x * blockDim.x + threadIdx.x;
  if (idx >= total) return;
  int i    = idx & 15;
  int lane = (idx >> 4) & 31;
  int tile = idx >> 9;            // kt*NT + nt
  int nt   = tile % NT;
  int kt   = tile / NT;
  int k = kt * 32 + ((lane & 16) ? 8 : 0) + (i & 7) + ((i & 8) ? 16 : 0);
  int n = nt * 16 + (lane & 15);
  Wp[idx] = (_Float16)W[(size_t)k * NOUT + n];
}

// ---------------- WMMA GEMM: out[M,NOUT] = A[M,K](f16) * Wp(packed f16) ----------------
// 1 wave = one 16-row M tile; 4 n-tiles per accumulator group share one A fragment.
// (Round-1 codegen: clause'd global_load_b128 pipelined against v_wmma chains.)
template<int K, int NOUT>
__global__ __launch_bounds__(256)
void gemm_wmma_f16(const _Float16* __restrict__ A,
                   const _Float16* __restrict__ Wp,
                   float* __restrict__ out) {
  constexpr int KT = K / 32;
  constexpr int NT = NOUT / 16;
  constexpr int NG = NOUT / 64;
  const int lane = threadIdx.x & 31;
  const int wave = threadIdx.x >> 5;
  const int mt   = blockIdx.x * 8 + wave;
  if (mt >= N_NODES / 16) return;              // wave-uniform: EXEC stays all-1s

  const int halfsel = (lane >> 4) & 1;         // 0: lanes 0-15, 1: lanes 16-31
  const size_t row  = (size_t)mt * 16 + (lane & 15);
  const _Float16* arow = A + row * K + halfsel * 8;
  float* orow = out + ((size_t)mt * 16 + halfsel * 8) * NOUT + (lane & 15);

  for (int g = 0; g < NG; ++g) {
    v8f acc[4];
    #pragma unroll
    for (int j = 0; j < 4; ++j)
      #pragma unroll
      for (int t = 0; t < 8; ++t) acc[j][t] = 0.0f;

    #pragma unroll
    for (int kt = 0; kt < KT; ++kt) {
      // A fragment: 16-bit 16x32 layout -> two contiguous 16B chunks per lane
      union { v16h v; v8h h[2]; } a;
      a.h[0] = *(const v8h*)(arow + kt * 32);
      a.h[1] = *(const v8h*)(arow + kt * 32 + 16);
      const _Float16* bp = Wp + ((size_t)(kt * NT + g * 4) * 32 + lane) * 16;
      #pragma unroll
      for (int j = 0; j < 4; ++j) {
        v16h b = *(const v16h*)(bp + (size_t)j * 512);
        acc[j] = __builtin_amdgcn_wmma_f32_16x16x32_f16(
            false, a.v, false, b, (short)0, acc[j], false, false);
      }
    }

    // C/D layout: VGPR r -> row (r + 8*halfsel), col = lane&15 within the n-tile
    #pragma unroll
    for (int j = 0; j < 4; ++j) {
      const int ncol = (g * 4 + j) * 16;
      #pragma unroll
      for (int r = 0; r < 8; ++r)
        orow[(size_t)r * NOUT + ncol] = acc[j][r];
    }
  }
}

// ---------------- CSR-by-dst build (integer atomics only, ~0.6M total) ----------------
__global__ void hist_dst(const int* __restrict__ dst, int* __restrict__ cnt, int nE) {
  int i = blockIdx.x * blockDim.x + threadIdx.x;
  if (i < nE) atomicAdd(&cnt[dst[i]], 1);
}

// Single-block exclusive scan: cntcur[i] (counts) -> offs[i] (+ cntcur rewritten to
// the same start offsets, to serve as the fill cursor). offs[n] = total.
__global__ __launch_bounds__(1024)
void scan_offsets(int* __restrict__ cntcur, int* __restrict__ offs, int n) {
  __shared__ int part[1024];
  const int t = threadIdx.x;
  const int chunk = (n + 1023) >> 10;
  const int lo = t * chunk;
  const int hi = min(lo + chunk, n);
  int s = 0;
  for (int i = lo; i < hi; ++i) s += cntcur[i];
  part[t] = s;
  __syncthreads();
  for (int d = 1; d < 1024; d <<= 1) {
    int v = (t >= d) ? part[t - d] : 0;
    __syncthreads();
    if (t >= d) part[t] += v;
    __syncthreads();
  }
  int run = (t == 0) ? 0 : part[t - 1];
  for (int i = lo; i < hi; ++i) {
    int c = cntcur[i];
    offs[i]   = run;
    cntcur[i] = run;
    run += c;
  }
  if (t == 0) offs[n] = part[1023];
}

__global__ void csr_fill(const int* __restrict__ src, const int* __restrict__ dst,
                         const float* __restrict__ val, int* __restrict__ cursor,
                         int* __restrict__ csr_src, float* __restrict__ csr_val, int nE) {
  int i = blockIdx.x * blockDim.x + threadIdx.x;
  if (i >= nE) return;
  int d = dst[i];
  int p = atomicAdd(&cursor[d], 1);
  csr_src[p] = src[i];
  csr_val[p] = val[i];
}

// ---------------- fused gather-aggregate + bias (+ReLU+f16 cast) ----------------
// thread t -> (node n, float4 chunk c). Consecutive lanes = consecutive chunks of the
// same node: (src,val) loads are wave-broadcast, support gathers are 128B-coalesced.
template<int D, bool FINAL>
__global__ void gather_bias_act(const float* __restrict__ sup,
                                const int* __restrict__ csr_src,
                                const float* __restrict__ csr_val,
                                const int* __restrict__ offs,
                                const float* __restrict__ bias,
                                _Float16* __restrict__ h16,
                                float* __restrict__ outf) {
  constexpr int C = D / 4;
  int t = blockIdx.x * blockDim.x + threadIdx.x;
  if (t >= N_NODES * C) return;
  int n = t / C;                               // shift (C is pow2)
  int c = (t - n * C) * 4;
  const int j1 = offs[n + 1];
  float a0 = 0.f, a1 = 0.f, a2 = 0.f, a3 = 0.f;
  for (int j = offs[n]; j < j1; ++j) {
    int s   = csr_src[j];
    float v = csr_val[j];
    vf4 m = *(const vf4*)(sup + (size_t)s * D + c);
    a0 = fmaf(v, m.x, a0);
    a1 = fmaf(v, m.y, a1);
    a2 = fmaf(v, m.z, a2);
    a3 = fmaf(v, m.w, a3);
  }
  a0 += bias[c];     a1 += bias[c + 1];
  a2 += bias[c + 2]; a3 += bias[c + 3];
  if (FINAL) {
    vf4 o = {a0, a1, a2, a3};
    ((vf4*)outf)[t] = o;
  } else {
    a0 = fmaxf(a0, 0.f); a1 = fmaxf(a1, 0.f);
    a2 = fmaxf(a2, 0.f); a3 = fmaxf(a3, 0.f);
    v4h h;
    h.x = (_Float16)a0; h.y = (_Float16)a1;
    h.z = (_Float16)a2; h.w = (_Float16)a3;
    ((v4h*)h16)[t] = h;
  }
}

// ---------------- driver ----------------
extern "C" void kernel_launch(void* const* d_in, const int* in_sizes, int n_in,
                              void* d_out, int out_size, void* d_ws, size_t ws_size,
                              hipStream_t stream) {
  const float* x  = (const float*)d_in[0];
  const float* W1 = (const float*)d_in[1];
  const float* b1 = (const float*)d_in[2];
  const float* W2 = (const float*)d_in[3];
  const float* b2 = (const float*)d_in[4];
  const float* W3 = (const float*)d_in[5];
  const float* b3 = (const float*)d_in[6];
  const float* ev = (const float*)d_in[7];
  const int*   es = (const int*)d_in[8];
  const int*   ed = (const int*)d_in[9];
  float* out = (float*)d_out;

  char* ws = (char*)d_ws;
  _Float16* bufA    = (_Float16*)(ws + OFF_A);
  float*    sup     = (float*)(ws + OFF_B);
  int*      offs    = (int*)(ws + OFF_OFFS);
  int*      cntcur  = (int*)(ws + OFF_CNT);
  int*      csr_src = (int*)(ws + OFF_CSRS);
  float*    csr_val = (float*)(ws + OFF_CSRV);
  _Float16* w1p     = (_Float16*)(ws + OFF_W1);
  _Float16* w2p     = (_Float16*)(ws + OFF_W2);
  _Float16* w3p     = (_Float16*)(ws + OFF_W3);

  const int TB = 256;
  const int gemmBlocks = (N_NODES / 16 + 7) / 8;   // 8 waves (M-tiles) per block

  // ---- prep: cast x, pack weights, build dst-CSR ----
  {
    int n4 = N_NODES * NFEAT / 4;
    cast_f32_to_f16_x4<<<(n4 + TB - 1) / TB, TB, 0, stream>>>(x, bufA, n4);
    int t1 = NFEAT * HID1;
    pack_w<<<(t1 + TB - 1) / TB, TB, 0, stream>>>(W1, w1p, HID1 / 16, HID1, t1);
    int t2 = HID1 * HID2;
    pack_w<<<(t2 + TB - 1) / TB, TB, 0, stream>>>(W2, w2p, HID2 / 16, HID2, t2);
    int t3 = HID2 * NHID;
    pack_w<<<(t3 + TB - 1) / TB, TB, 0, stream>>>(W3, w3p, NHID / 16, NHID, t3);

    int z4 = (int)(SZ_OFFS / 16);                 // zero padded count region
    zero_f32_x4<<<(z4 + TB - 1) / TB, TB, 0, stream>>>((float*)cntcur, z4);
    hist_dst<<<(N_EDGES + TB - 1) / TB, TB, 0, stream>>>(ed, cntcur, N_EDGES);
    scan_offsets<<<1, 1024, 0, stream>>>(cntcur, offs, N_NODES);
    csr_fill<<<(N_EDGES + TB - 1) / TB, TB, 0, stream>>>(es, ed, ev, cntcur,
                                                         csr_src, csr_val, N_EDGES);
  }

  // ---- layer 1: 512 -> 256, ReLU, cast f16 ----
  {
    gemm_wmma_f16<NFEAT, HID1><<<gemmBlocks, TB, 0, stream>>>(bufA, w1p, sup);
    int nt = N_NODES * (HID1 / 4);
    gather_bias_act<HID1, false><<<(nt + TB - 1) / TB, TB, 0, stream>>>(
        sup, csr_src, csr_val, offs, b1, bufA, nullptr);
  }

  // ---- layer 2: 256 -> 128, ReLU, cast f16 ----
  {
    gemm_wmma_f16<HID1, HID2><<<gemmBlocks, TB, 0, stream>>>(bufA, w2p, sup);
    int nt = N_NODES * (HID2 / 4);
    gather_bias_act<HID2, false><<<(nt + TB - 1) / TB, TB, 0, stream>>>(
        sup, csr_src, csr_val, offs, b2, bufA, nullptr);
  }

  // ---- layer 3: 128 -> 64, bias only, fp32 straight into d_out ----
  {
    gemm_wmma_f16<HID2, NHID><<<gemmBlocks, TB, 0, stream>>>(bufA, w3p, sup);
    int nt = N_NODES * (NHID / 4);
    gather_bias_act<NHID, true><<<(nt + TB - 1) / TB, TB, 0, stream>>>(
        sup, csr_src, csr_val, offs, b3, nullptr, out);
  }
}
